// LSTM_69501160783998
// MI455X (gfx1250) — matrix operand order repaired and checked by reading, not compiled, not run on hardware
//
#include <hip/hip_runtime.h>
#include <hip/hip_bf16.h>
#include <math.h>

// Problem constants (match reference)
#define IC 512
#define HC 512
#define G4 2048    // 4*HC
#define KTOT 1024  // IC+HC
#define TSEQ 512
#define BS 64
#define NLAYER 2
#define NWG 32
#define NTHREADS 256
#define NKC (KTOT / 32)

typedef __attribute__((ext_vector_type(16))) __bf16 v16bf;
typedef __attribute__((ext_vector_type(8)))  __bf16 v8bf;
typedef __attribute__((ext_vector_type(8)))  float  v8f;

union BF16Frag { v16bf v; v8bf h[2]; };

// ---------------- prologue: transpose weights to [N][K] bf16 ----------------
__global__ void lstm_init_weights(const float* __restrict__ W0,
                                  const float* __restrict__ W1,
                                  __bf16* __restrict__ Wt0,
                                  __bf16* __restrict__ Wt1) {
  size_t idx = (size_t)blockIdx.x * blockDim.x + threadIdx.x;
  size_t total = (size_t)G4 * KTOT;
  if (idx >= 2 * total) return;
  int lay = idx >= total;
  size_t r = idx - (size_t)lay * total;
  int n = (int)(r / KTOT);
  int k = (int)(r % KTOT);
  const float* W = lay ? W1 : W0;            // [K][N] row-major
  __bf16* Wt = lay ? Wt1 : Wt0;              // [N][K]
  Wt[(size_t)n * KTOT + k] = (__bf16)W[(size_t)k * G4 + n];
}

// ---------------- prologue: broadcast h0 into h_state, reset barrier -------
__global__ void lstm_init_state(const float* __restrict__ h0,
                                float* __restrict__ h_state,
                                unsigned* __restrict__ bar) {
  int idx = blockIdx.x * blockDim.x + threadIdx.x;
  if (idx < NLAYER * BS * HC) {
    int lay = idx / (BS * HC);
    int j = idx % HC;
    h_state[idx] = h0[lay * HC + j];         // h0 is [L,1,HC], broadcast over batch
  }
  if (idx < 2) bar[idx] = 0;
}

// ---------------- grid-wide barrier (32 resident WGs) ----------------------
__device__ inline void grid_barrier(unsigned* bar) {
  __syncthreads();
  if (threadIdx.x == 0) {
    __threadfence();
    volatile unsigned* gen = bar + 1;
    unsigned g = *gen;
    if (atomicAdd(bar, 1u) == NWG - 1) {
      *bar = 0;
      __threadfence();
      atomicAdd((unsigned*)gen, 1u);
    } else {
      while (*gen == g) { __builtin_amdgcn_s_sleep(1); }
    }
  }
  __syncthreads();
}

// ---------------- hardware transcendentals (CDNA5 V_TANH_F32) ---------------
__device__ inline float fast_tanh(float x) {
#if __has_builtin(__builtin_amdgcn_tanhf)
  return __builtin_amdgcn_tanhf(x);
#elif __has_builtin(__builtin_amdgcn_tanh_f32)
  return __builtin_amdgcn_tanh_f32(x);
#else
  return tanhf(x);
#endif
}
__device__ inline float fast_sigmoid(float x) {
  return __builtin_fmaf(0.5f, fast_tanh(0.5f * x), 0.5f);
}

// ---- cooperative f32->bf16 stage of one 64x32 A chunk into LDS -------------
__device__ inline void stage_a_chunk(const float* __restrict__ x_t,
                                     const float* __restrict__ h_state,
                                     int layer, int kglob,
                                     __bf16* __restrict__ buf, int tid) {
  const float* baseA;
  if (kglob < IC) {
    baseA = (layer == 0) ? (x_t + kglob) : (h_state + kglob);   // layer1 input = h0_t
  } else {
    baseA = h_state + (size_t)layer * BS * HC + (kglob - IC);   // recurrent h
  }
  const int row = tid >> 2;
  const int kq  = (tid & 3) * 8;
  const float* p = baseA + (size_t)row * 512 + kq;
  float4 f0 = *(const float4*)(p);
  float4 f1 = *(const float4*)(p + 4);
  v8bf pk;
  pk[0] = (__bf16)f0.x; pk[1] = (__bf16)f0.y;
  pk[2] = (__bf16)f0.z; pk[3] = (__bf16)f0.w;
  pk[4] = (__bf16)f1.x; pk[5] = (__bf16)f1.y;
  pk[6] = (__bf16)f1.z; pk[7] = (__bf16)f1.w;
  *(v8bf*)&buf[row * 32 + kq] = pk;
}

// ---------------- persistent cooperative LSTM -------------------------------
__global__ void __launch_bounds__(NTHREADS)
lstm_persistent(const float* __restrict__ x,
                const float* __restrict__ b0,
                const float* __restrict__ b1,
                const float* __restrict__ c0in,
                const __bf16* __restrict__ Wt0,
                const __bf16* __restrict__ Wt1,
                float* h_state,          // [2][BS][HC]
                unsigned* bar,
                float* out) {            // [T][BS][HC] ++ hs[2][BS][HC] ++ cs[2][BS][HC]
  __shared__ __bf16 a_lds[2][BS * 32];           // double-buffered A chunk (2 x 4 KB)
  __shared__ float  z_lds[BS * 64];              // gate pre-activations (16 KB)
  __shared__ float  c_lds[NLAYER * BS * 16];     // persistent cell state (8 KB)

  const int tid   = threadIdx.x;
  const int wg    = blockIdx.x;
  const int lane  = tid & 31;
  const int wave  = tid >> 5;
  const int gate  = wave & 3;       // f,i,o,g
  const int mhalf = wave >> 2;      // which 32 batch rows
  const int j0    = wg * 16;        // this WG's hidden-unit slice
  const int nlo   = lane & 15;
  const int khi   = lane >> 4;

  // load initial cell state (broadcast over batch)
  for (int e = tid; e < NLAYER * BS * 16; e += NTHREADS) {
    int lay = e >> 10;
    int j = e & 15;
    c_lds[e] = c0in[lay * HC + j0 + j];
  }
  __syncthreads();

  const int ncol = gate * HC + j0 + nlo;         // this lane's output column
  // t-invariant hoists: weight column pointers + biases for both layers
  const __bf16* wcol_l[NLAYER] = { Wt0 + (size_t)ncol * KTOT,
                                   Wt1 + (size_t)ncol * KTOT };
  const float bcol_l[NLAYER] = { b0[ncol], b1[ncol] };
  const int m_base = (mhalf * 2) * 16 + nlo;
  const int klo = khi * 8;

#pragma unroll 1
  for (int t = 0; t < TSEQ; ++t) {
    const float* x_t = x + (size_t)t * BS * IC;
#pragma unroll 1
    for (int layer = 0; layer < NLAYER; ++layer) {
      const __bf16* wcol = wcol_l[layer];

      v8f acc0 = {};
      v8f acc1 = {};

      // software pipeline: stage chunk 0, then overlap stage(k+1) with wmma(k)
      stage_a_chunk(x_t, h_state, layer, 0, a_lds[0], tid);
      __syncthreads();

#pragma unroll 2
      for (int kc = 0; kc < NKC; ++kc) {
        const __bf16* abuf = a_lds[kc & 1];

        // ---- B fragment: K 8h..8h+7 and 16+8h..16+8h+7 of this lane's column
        BF16Frag bfr;
        const __bf16* bp = wcol + kc * 32 + khi * 8;
        bfr.h[0] = *(const v8bf*)(bp);
        bfr.h[1] = *(const v8bf*)(bp + 16);

        // ---- A fragments for two M tiles (ISA 16-bit A layout)
        BF16Frag a0, a1;
        {
          const __bf16* ap = &abuf[m_base * 32 + klo];
          a0.h[0] = *(const v8bf*)(ap);
          a0.h[1] = *(const v8bf*)(ap + 16);
          ap += 16 * 32;
          a1.h[0] = *(const v8bf*)(ap);
          a1.h[1] = *(const v8bf*)(ap + 16);
        }

        // ---- stage next chunk into the other buffer (overlaps WMMA) ----
        if (kc + 1 < NKC)
          stage_a_chunk(x_t, h_state, layer, (kc + 1) * 32,
                        a_lds[(kc + 1) & 1], tid);

        acc0 = __builtin_amdgcn_wmma_f32_16x16x32_bf16(
            false, a0.v, false, bfr.v, (short)0, acc0, false, false);
        acc1 = __builtin_amdgcn_wmma_f32_16x16x32_bf16(
            false, a1.v, false, bfr.v, (short)0, acc1, false, false);
        __syncthreads();
      }

      // ---- bias + spill gate pre-activations to LDS ----
      const float bcol = bcol_l[layer];
#pragma unroll
      for (int v = 0; v < 8; ++v) {
        int m0 = m_base - nlo + khi * 8 + v;
        z_lds[m0 * 64 + gate * 16 + nlo]        = acc0[v] + bcol;
        z_lds[(m0 + 16) * 64 + gate * 16 + nlo] = acc1[v] + bcol;
      }
      __syncthreads();

      // ---- LSTM cell elementwise (WG-local: 64 batch x 16 units) ----
      {
        const int j  = tid & 15;
        const int m4 = tid >> 4;
#pragma unroll
        for (int mm = 0; mm < 4; ++mm) {
          int m = m4 * 4 + mm;
          float zf = z_lds[m * 64 + 0 * 16 + j] + 1.0f;   // forget bias
          float zi = z_lds[m * 64 + 1 * 16 + j];
          float zo = z_lds[m * 64 + 2 * 16 + j];
          float zg = z_lds[m * 64 + 3 * 16 + j];
          float f  = fast_sigmoid(zf);
          float i  = fast_sigmoid(zi);
          float o  = fast_sigmoid(zo);
          float gg = fast_tanh(zg);
          float c  = c_lds[layer * (BS * 16) + m * 16 + j];
          c = c * f + gg * i;
          float h = o * fast_tanh(c);
          c_lds[layer * (BS * 16) + m * 16 + j] = c;
          h_state[(size_t)layer * BS * HC + (size_t)m * HC + j0 + j] = h;
          if (layer == 1)
            out[((size_t)t * BS + m) * HC + j0 + j] = h;
        }
      }
      grid_barrier(bar);
    }
  }

  // ---- final hT / cT for both layers (own slice only) ----
  for (int e = tid; e < NLAYER * BS * 16; e += NTHREADS) {
    int lay = e >> 10;
    int r = e & 1023;
    int m = r >> 4;
    int j = r & 15;
    size_t base = (size_t)TSEQ * BS * HC;
    out[base + (size_t)lay * BS * HC + (size_t)m * HC + j0 + j] =
        h_state[(size_t)lay * BS * HC + (size_t)m * HC + j0 + j];
    out[base + (size_t)NLAYER * BS * HC + (size_t)lay * BS * HC +
        (size_t)m * HC + j0 + j] = c_lds[e];
  }
}

// ---------------------------------------------------------------------------
extern "C" void kernel_launch(void* const* d_in, const int* in_sizes, int n_in,
                              void* d_out, int out_size, void* d_ws, size_t ws_size,
                              hipStream_t stream) {
  const float* x  = (const float*)d_in[0];
  const float* W0 = (const float*)d_in[1];
  const float* b0 = (const float*)d_in[2];
  const float* W1 = (const float*)d_in[3];
  const float* b1 = (const float*)d_in[4];
  const float* h0 = (const float*)d_in[5];
  const float* c0 = (const float*)d_in[6];
  float* out = (float*)d_out;

  char* ws = (char*)d_ws;
  const size_t wbytes = (size_t)G4 * KTOT * sizeof(__bf16);   // 4 MB per layer
  __bf16*   Wt0     = (__bf16*)(ws);
  __bf16*   Wt1     = (__bf16*)(ws + wbytes);
  float*    h_state = (float*)(ws + 2 * wbytes);
  unsigned* bar     = (unsigned*)(ws + 2 * wbytes +
                                  (size_t)NLAYER * BS * HC * sizeof(float));

  {
    size_t total = 2ull * G4 * KTOT;
    int blocks = (int)((total + 255) / 256);
    lstm_init_weights<<<blocks, 256, 0, stream>>>(W0, W1, Wt0, Wt1);
  }
  {
    int n = NLAYER * BS * HC;
    lstm_init_state<<<(n + 255) / 256, 256, 0, stream>>>(h0, h_state, bar);
  }
  lstm_persistent<<<NWG, NTHREADS, 0, stream>>>(x, b0, b1, c0, Wt0, Wt1,
                                                h_state, bar, out);
}
